// Model_35124242547372
// MI455X (gfx1250) — matrix operand order repaired and checked by reading, not compiled, not run on hardware
//
#include <hip/hip_runtime.h>
#include <hip/hip_bf16.h>

// ---------------- problem constants ----------------
constexpr int Bc = 512, Tc = 256, Dc = 64, Hc = 128;
constexpr int TD = Tc * Dc;          // 16384
constexpr int GN = 4 * Hc;           // 512 gate cols
constexpr int GK = 2 * Dc + Hc;      // 256 gate K (inp=128 | h=128)

typedef __attribute__((ext_vector_type(16))) _Float16 v16h;
typedef __attribute__((ext_vector_type(8)))  _Float16 v8h;
typedef __attribute__((ext_vector_type(8)))  float    v8f;

// ---------------- shared memory layout (one 16-row batch tile) ------------
struct __align__(16) Smem {
    // f16 weight matrices, B-layout [N][K] (== row-major W for y = x @ W.T)
    _Float16 WdhB[Hc * Dc];      // 128x64
    _Float16 WhrB[Dc * Hc];      // 64x128
    _Float16 WfrB[Dc * Dc];      // 64x64 (zero diag)
    _Float16 WwcB[Dc * 2 * Dc];  // 64x128
    // f16 activation tiles (A-layout [16][K], K contiguous)
    _Float16 dA[16 * Dc];
    _Float16 mA[16 * Dc];
    _Float16 xcA[16 * Dc];
    _Float16 ccA[16 * Dc];
    _Float16 gxm[16 * 2 * Dc];   // [gamma_x | m]
    _Float16 hA[16 * Hc];        // decayed h (f16)
    // f32 state / staging
    float bdh[Hc], wdxd[Dc], bdx[Dc], bhr[Dc], bfr[Dc], bwc[Dc], bg[GN];
    float xs[16 * Dc], ms[16 * Dc];
    float hs[16 * Hc], cs[16 * Hc];
    float red[128];
};

// ---------------- WMMA fragment helpers (CDNA5 16x16x32 f16 layouts) ------
__device__ __forceinline__ v16h fragA(const _Float16* A, int ldk, int koff, int lane) {
    // lane<16: row=lane, K = koff + {0..7, 16..23}; lane>=16: K shifted by 8
    const _Float16* p = A + (lane & 15) * ldk + koff + ((lane >> 4) << 3);
    v8h lo = *(const v8h*)(p);
    v8h hi = *(const v8h*)(p + 16);
    v16h a;
#pragma unroll
    for (int j = 0; j < 8; ++j) { a[j] = lo[j]; a[j + 8] = hi[j]; }
    return a;
}

__device__ __forceinline__ v16h fragB(const _Float16* Bm, int ldk, int noff, int koff, int lane) {
    // N = noff + (lane&15); K = koff + 8*(lane>=16) + {0..7, 16..23}, in [N][K]
    const _Float16* p = Bm + (noff + (lane & 15)) * ldk + koff + ((lane >> 4) << 3);
    v8h lo = *(const v8h*)(p);
    v8h hi = *(const v8h*)(p + 16);
    v16h b;
#pragma unroll
    for (int j = 0; j < 8; ++j) { b[j] = lo[j]; b[j + 8] = hi[j]; }
    return b;
}

__device__ __forceinline__ v8f wmma32(v16h a, v16h b, v8f c) {
    return __builtin_amdgcn_wmma_f32_16x16x32_f16(false, a, false, b, (short)0, c, false, false);
}

__device__ __forceinline__ float sigf(float x) { return 1.0f / (1.0f + __expf(-x)); }

// ---------------- kernel 1: per-timestep 1/(sum(mask)+1e-5), zero the loss -
__global__ void msum_kernel(const float* __restrict__ masks,
                            float* __restrict__ inv_msum,
                            float* __restrict__ loss_out) {
    __shared__ float red[256];
    int t = blockIdx.x;
    float s = 0.f;
    for (int i = threadIdx.x; i < Bc * Dc; i += 256) {
        int b = i >> 6, d = i & 63;
        s += masks[(size_t)b * TD + (size_t)t * Dc + d];
    }
    red[threadIdx.x] = s;
    __syncthreads();
    for (int off = 128; off > 0; off >>= 1) {
        if (threadIdx.x < off) red[threadIdx.x] += red[threadIdx.x + off];
        __syncthreads();
    }
    if (threadIdx.x == 0) {
        inv_msum[t] = 1.0f / (red[0] + 1e-5f);
        if (t == 0) loss_out[0] = 0.0f;
    }
}

// ---------------- kernel 2: fuse+convert gate weights to f16 [512][256] ----
__global__ void wcvt_kernel(const float* __restrict__ W_ih, const float* __restrict__ W_hh,
                            _Float16* __restrict__ W2) {
    int i = blockIdx.x * blockDim.x + threadIdx.x;   // 0 .. 512*256-1
    if (i < GN * GK) {
        int n = i >> 8, k = i & 255;
        float v = (k < 128) ? W_ih[n * 128 + k] : W_hh[n * 128 + (k - 128)];
        W2[i] = (_Float16)v;
    }
}

// ---------------- kernel 3: full 256-step recurrence, 16 batch rows / WG ---
__global__ __launch_bounds__(128, 1) void brits_kernel(
    const float* __restrict__ x_t, const float* __restrict__ masks, const float* __restrict__ deltas,
    const float* __restrict__ Wdh, const float* __restrict__ bdh,
    const float* __restrict__ Wdx, const float* __restrict__ bdx,
    const float* __restrict__ Whr, const float* __restrict__ bhr,
    const float* __restrict__ Wfr, const float* __restrict__ bfr,
    const float* __restrict__ Wwc, const float* __restrict__ bwc,
    const float* __restrict__ b_ih, const float* __restrict__ b_hh,
    const _Float16* __restrict__ W2, const float* __restrict__ inv_msum,
    float* __restrict__ loss_out, float* __restrict__ imp_out) {
    extern __shared__ char smem_raw[];
    Smem& s = *reinterpret_cast<Smem*>(smem_raw);

    const int tid  = threadIdx.x;
    const int lane = tid & 31;
    const int w    = tid >> 5;      // wave 0..3
    const int nl   = lane & 15;
    const int hi   = lane >> 4;
    const int b0   = blockIdx.x * 16;

    // ---- stage small weights (f16) + biases, zero h/c ----
    for (int i = tid; i < Hc * Dc; i += 128) s.WdhB[i] = (_Float16)Wdh[i];
    for (int i = tid; i < Dc * Hc; i += 128) s.WhrB[i] = (_Float16)Whr[i];
    for (int i = tid; i < Dc * Dc; i += 128) {
        int r = i >> 6, c = i & 63;
        s.WfrB[i] = (_Float16)((r == c) ? 0.0f : Wfr[i]);   // zero-diagonal
    }
    for (int i = tid; i < Dc * 2 * Dc; i += 128) s.WwcB[i] = (_Float16)Wwc[i];
    if (tid < Hc) s.bdh[tid] = bdh[tid];
    if (tid < Dc) {
        s.wdxd[tid] = Wdx[tid * Dc + tid];                  // diagonal decay
        s.bdx[tid] = bdx[tid]; s.bhr[tid] = bhr[tid];
        s.bfr[tid] = bfr[tid]; s.bwc[tid] = bwc[tid];
    }
    for (int i = tid; i < GN; i += 128) s.bg[i] = b_ih[i] + b_hh[i];
    for (int i = tid; i < 16 * Hc; i += 128) { s.hs[i] = 0.f; s.cs[i] = 0.f; }
    __syncthreads();

    float loss_local = 0.f;

    for (int t = 0; t < Tc; ++t) {
        // ---- phase 1: load x/m/d tile, gamma_x, build [gamma_x|m] f16 ----
#pragma unroll
        for (int j = 0; j < 8; ++j) {
            int e = tid + 128 * j;
            int row = e >> 6, col = e & 63;
            size_t g = (size_t)(b0 + row) * TD + (size_t)t * Dc + col;
            float xv = x_t[g], mv = masks[g], dv = deltas[g];
            s.xs[e] = xv; s.ms[e] = mv;
            s.mA[e] = (_Float16)mv; s.dA[e] = (_Float16)dv;
            float gx = __expf(-fmaxf(0.f, dv * s.wdxd[col] + s.bdx[col]));
            s.gxm[row * 128 + col] = (_Float16)gx;
            s.gxm[row * 128 + 64 + col] = (_Float16)mv;
            if (t + 1 < Tc) {   // prefetch next step's tiles into cache
                __builtin_prefetch(&x_t[g + Dc], 0, 3);
                __builtin_prefetch(&masks[g + Dc], 0, 3);
                __builtin_prefetch(&deltas[g + Dc], 0, 3);
            }
        }
        __syncthreads();                                        // s1

        // ---- phase 2: gamma_h = exp(-relu(d@Wdh.T+bdh)); h *= gamma_h ----
#pragma unroll
        for (int tt = 0; tt < 2; ++tt) {
            int noff = w * 32 + tt * 16;
            v8f acc = {};
            acc = wmma32(fragA(s.dA, Dc, 0,  lane), fragB(s.WdhB, Dc, noff, 0,  lane), acc);
            acc = wmma32(fragA(s.dA, Dc, 32, lane), fragB(s.WdhB, Dc, noff, 32, lane), acc);
#pragma unroll
            for (int v = 0; v < 8; ++v) {
                int row = v + 8 * hi, col = noff + nl;
                float gh = __expf(-fmaxf(0.f, acc[v] + s.bdh[col]));
                float hv = s.hs[row * Hc + col] * gh;
                s.hs[row * Hc + col] = hv;
                s.hA[row * Hc + col] = (_Float16)hv;
            }
        }
        __syncthreads();                                        // s2

        // ---- phase 3: x_h = h@Whr.T + bhr ; loss1 ; x_c ----
        float xr[8], mr[8], xhr[8];
        float l1 = 0.f, l2 = 0.f, l3 = 0.f;
        {
            int noff = w * 16;
            v8f acc = {};
#pragma unroll
            for (int kf = 0; kf < 4; ++kf)
                acc = wmma32(fragA(s.hA, Hc, kf * 32, lane),
                             fragB(s.WhrB, Hc, noff, kf * 32, lane), acc);
#pragma unroll
            for (int v = 0; v < 8; ++v) {
                int row = v + 8 * hi, col = noff + nl, e = row * Dc + col;
                float xhv = acc[v] + s.bhr[col];
                float xv = s.xs[e], mv = s.ms[e];
                xr[v] = xv; mr[v] = mv; xhr[v] = xhv;
                l1 += fabsf(xv - xhv) * mv;
                s.xcA[e] = (_Float16)(mv * xv + (1.f - mv) * xhv);
            }
        }

        // ---- gates (pipelined): m-part (K 64..127) + h-part (K 128..255)
        //      are independent of the x_h/z_h/alpha chain; issue them now.
        //      Wave w owns cols [32w,32w+32) of ALL four gates so the LSTM
        //      update can be done entirely in registers afterwards.
        v8f gacc[2][4];
#pragma unroll
        for (int tt = 0; tt < 2; ++tt)
#pragma unroll
            for (int g = 0; g < 4; ++g) {
                int noff = 128 * g + 32 * w + 16 * tt;
                v8f a = {};
                a = wmma32(fragA(s.mA, Dc, 0,  lane), fragB(W2, GK, noff, 64, lane), a);
                a = wmma32(fragA(s.mA, Dc, 32, lane), fragB(W2, GK, noff, 96, lane), a);
#pragma unroll
                for (int kf = 0; kf < 4; ++kf)
                    a = wmma32(fragA(s.hA, Hc, kf * 32, lane),
                               fragB(W2, GK, noff, 128 + kf * 32, lane), a);
                gacc[tt][g] = a;
            }
        __syncthreads();                                        // s3a

        // ---- phase 4+5: z_h, alpha, c_h, loss2/3, imputation c_c ----
        {
            int noff = w * 16;
            v8f az = {};
            az = wmma32(fragA(s.xcA, Dc, 0,  lane), fragB(s.WfrB, Dc, noff, 0,  lane), az);
            az = wmma32(fragA(s.xcA, Dc, 32, lane), fragB(s.WfrB, Dc, noff, 32, lane), az);
            v8f aa = {};
#pragma unroll
            for (int kf = 0; kf < 4; ++kf)
                aa = wmma32(fragA(s.gxm, 2 * Dc, kf * 32, lane),
                            fragB(s.WwcB, 2 * Dc, noff, kf * 32, lane), aa);
#pragma unroll
            for (int v = 0; v < 8; ++v) {
                int row = v + 8 * hi, col = noff + nl, e = row * Dc + col;
                float zh = az[v] + s.bfr[col];
                float al = aa[v] + s.bwc[col];
                float xv = xr[v], mv = mr[v], xhv = xhr[v];
                l2 += fabsf(xv - zh) * mv;
                float ch = al * zh + (1.f - al) * xhv;
                l3 += fabsf(xv - ch) * mv;
                float cc = mv * xv + (1.f - mv) * ch;
                imp_out[(size_t)(b0 + row) * TD + (size_t)t * Dc + col] = cc;
                s.ccA[e] = (_Float16)cc;
            }
        }
        loss_local += (l1 + l2 + l3) * inv_msum[t];
        __syncthreads();                                        // s3b

        // ---- gates finish: c_c part (K 0..63), then LSTM update in regs ---
#pragma unroll
        for (int tt = 0; tt < 2; ++tt)
#pragma unroll
            for (int g = 0; g < 4; ++g) {
                int noff = 128 * g + 32 * w + 16 * tt;
                v8f a = gacc[tt][g];
                a = wmma32(fragA(s.ccA, Dc, 0,  lane), fragB(W2, GK, noff, 0,  lane), a);
                a = wmma32(fragA(s.ccA, Dc, 32, lane), fragB(W2, GK, noff, 32, lane), a);
                gacc[tt][g] = a;
            }
#pragma unroll
        for (int tt = 0; tt < 2; ++tt) {
            int colb = 32 * w + 16 * tt + nl;     // col within [0,128)
#pragma unroll
            for (int v = 0; v < 8; ++v) {
                int row = v + 8 * hi, e = row * Hc + colb;
                float iv = gacc[tt][0][v] + s.bg[colb];
                float fv = gacc[tt][1][v] + s.bg[128 + colb];
                float gv = gacc[tt][2][v] + s.bg[256 + colb];
                float ov = gacc[tt][3][v] + s.bg[384 + colb];
                float cv = s.cs[e];
                float cn = sigf(fv) * cv + sigf(iv) * tanhf(gv);
                s.cs[e] = cn;
                s.hs[e] = sigf(ov) * tanhf(cn);
            }
        }
        // next-iteration barrier (s1) orders hs/cs for phase 2
    }

    // ---- block-wide loss reduction -> atomic add (reference returns loss/T)
    __syncthreads();
    s.red[tid] = loss_local;
    __syncthreads();
    for (int off = 64; off > 0; off >>= 1) {
        if (tid < off) s.red[tid] += s.red[tid + off];
        __syncthreads();
    }
    if (tid == 0) atomicAdd(loss_out, s.red[0] * (1.0f / (float)Tc));
}

// ---------------- host-side launch ----------------
extern "C" void kernel_launch(void* const* d_in, const int* in_sizes, int n_in,
                              void* d_out, int out_size, void* d_ws, size_t ws_size,
                              hipStream_t stream) {
    const float* x_t    = (const float*)d_in[0];
    const float* masks  = (const float*)d_in[1];
    const float* deltas = (const float*)d_in[2];
    const float* Wdh = (const float*)d_in[3];
    const float* bdh = (const float*)d_in[4];
    const float* Wdx = (const float*)d_in[5];
    const float* bdx = (const float*)d_in[6];
    const float* Whr = (const float*)d_in[7];
    const float* bhr = (const float*)d_in[8];
    const float* Wfr = (const float*)d_in[9];
    const float* bfr = (const float*)d_in[10];
    const float* Wwc = (const float*)d_in[11];
    const float* bwc = (const float*)d_in[12];
    const float* W_ih = (const float*)d_in[13];
    const float* W_hh = (const float*)d_in[14];
    const float* b_ih = (const float*)d_in[15];
    const float* b_hh = (const float*)d_in[16];

    float* out = (float*)d_out;              // [0]=loss, [1..]=imputations
    float* inv_msum = (float*)d_ws;          // 256 floats
    _Float16* W2 = (_Float16*)((char*)d_ws + 1024);   // 512x256 f16 = 256 KB

    msum_kernel<<<Tc, 256, 0, stream>>>(masks, inv_msum, out);
    wcvt_kernel<<<(GN * GK + 255) / 256, 256, 0, stream>>>(W_ih, W_hh, W2);

    (void)hipFuncSetAttribute((const void*)brits_kernel,
                              hipFuncAttributeMaxDynamicSharedMemorySize,
                              (int)sizeof(Smem));
    brits_kernel<<<Bc / 16, 128, sizeof(Smem), stream>>>(
        x_t, masks, deltas, Wdh, bdh, Wdx, bdx, Whr, bhr, Wfr, bfr,
        Wwc, bwc, b_ih, b_hh, W2, inv_msum, out, out + 1);
}
